// Attention_56736517980393
// MI455X (gfx1250) — matrix operand order repaired and checked by reading, not compiled, not run on hardware
//
#include <hip/hip_runtime.h>
#include <hip/hip_bf16.h>
#include <math.h>

// ---------------------------------------------------------------------------
// bf16 WMMA fused-transformer-block pipeline for gfx1250 (MI455X), v4.
//   - every GEMM through v_wmma_f32_16x16x32_bf16, f32 accumulation
//   - wave tile 32x64 (2x4 fragments) -> 16 WMMA per 64-wide K stage
//   - double-buffered ASYNC global->LDS staging (global_load_async_to_lds_b128
//     + s_wait_asynccnt) when the toolchain exposes it; falls back to
//     b128 load/ds_store staging otherwise
//   - fused bias/scale/exact-GELU/residual epilogues
// ---------------------------------------------------------------------------

typedef __bf16 bf16;
typedef __attribute__((ext_vector_type(16))) __bf16 v16bf;
typedef __attribute__((ext_vector_type(8)))  __bf16 v8bf;
typedef __attribute__((ext_vector_type(8)))  float  v8f;

#if __has_builtin(__builtin_amdgcn_global_load_async_to_lds_b128)
#define HAVE_ASYNC_LDS 1
typedef int v4i __attribute__((vector_size(16)));
typedef __attribute__((address_space(1))) v4i as1_v4i;  // global-space 128b
typedef __attribute__((address_space(3))) v4i as3_v4i;  // LDS-space 128b
#if __has_builtin(__builtin_amdgcn_s_wait_asynccnt)
#define WAIT_ASYNC(n) __builtin_amdgcn_s_wait_asynccnt(n)
#else
#define WAIT_ASYNC(n) asm volatile("s_wait_asynccnt %0" ::"n"(n) : "memory")
#endif
#else
#define HAVE_ASYNC_LDS 0
#endif

struct U128 { unsigned x, y, z, w; };

__device__ __forceinline__ v8bf zero8() {
  U128 z; z.x = z.y = z.z = z.w = 0u;
  return __builtin_bit_cast(v8bf, z);
}

__device__ __forceinline__ bf16 f2bf(float f) {
  union { float f; unsigned u; } x;
  x.f = f;
  unsigned u = x.u;
  unsigned r = u + 0x7FFFu + ((u >> 16) & 1u);  // round-to-nearest-even
  unsigned short h = (unsigned short)(r >> 16);
  return __builtin_bit_cast(bf16, h);
}

__global__ void cvt_f32_bf16_kernel(const float* __restrict__ in,
                                    bf16* __restrict__ out, int n) {
  int i = blockIdx.x * blockDim.x + threadIdx.x;
  if (i < n) out[i] = f2bf(in[i]);
}

// f32 [K][N] row-major  ->  bf16 [N][K] row-major (transposed), LDS-tiled.
__global__ __launch_bounds__(256) void cvtT_f32_bf16_kernel(
    const float* __restrict__ in, bf16* __restrict__ out, int Kd, int Nd) {
  __shared__ float t[32][33];
  const int k0 = blockIdx.y * 32, n0 = blockIdx.x * 32;
  const int tx = threadIdx.x & 31, ty = threadIdx.x >> 5;  // ty in 0..7
  for (int i = ty; i < 32; i += 8)
    t[i][tx] = in[(long long)(k0 + i) * Nd + n0 + tx];
  __syncthreads();
  for (int i = ty; i < 32; i += 8)
    out[(long long)(n0 + i) * Kd + k0 + tx] = f2bf(t[tx][i]);
}

// Tile geometry: 256 threads = 8 waves (4 along M x 2 along N).
// Block tile 128(M) x 128(N); wave tile 32x64 = 2x4 fragments; K stage 64.
#define TM 128
#define TN 128
#define TK 64

// 2 WMMA K-steps per stage: 16 v_wmma per wave per stage.
#define COMPUTE_STAGE(Asb, Bsb)                                              \
  for (int ks = 0; ks < 2; ++ks) {                                           \
    union { v16bf v; v8bf h[2]; } afrag[2], bfrag[4];                        \
    const int kb = ks * 32;                                                  \
    for (int f = 0; f < 2; ++f) {                                            \
      const bf16* ap = &(Asb)[wm + f * 16 + lm][kb];                         \
      afrag[f].h[0] = *(const v8bf*)(ap + lk8);                              \
      afrag[f].h[1] = *(const v8bf*)(ap + lk8 + 16);                         \
    }                                                                        \
    for (int f = 0; f < 4; ++f) {                                            \
      const bf16* bp = &(Bsb)[wn + f * 16 + lm][kb];                         \
      bfrag[f].h[0] = *(const v8bf*)(bp + lk8);                              \
      bfrag[f].h[1] = *(const v8bf*)(bp + lk8 + 16);                         \
    }                                                                        \
    for (int fm = 0; fm < 2; ++fm)                                           \
      for (int fn = 0; fn < 4; ++fn)                                         \
        acc[fm][fn] = __builtin_amdgcn_wmma_f32_16x16x32_bf16(               \
            false, afrag[fm].v, false, bfrag[fn].v,                          \
            (short)0, acc[fm][fn], false, false);                            \
  }

// Batched strided GEMM, all operands contiguous along K:
//   A  indexed [offA(z) + m*sAm + k],  Bt indexed [offB(z) + n*sBn + k]
//   z = (b,h): off = b*sXb + h*sXh
// Epilogue: v = (acc + bias[n]) * alpha; optional exact GELU;
//   f32 path:  ci = offC + gm*ldc + gn; += res1/res2; outF[ci] = v
//   bf16 path: ci = offC + gm*sCbm + gn*sCbn (vectorized when sCbm == 1)
__global__ __launch_bounds__(256) void gemm_bf16_kernel(
    const bf16* __restrict__ A, int sAm, long long sAb, int sAh,
    const bf16* __restrict__ Bt, int sBn, long long sBb, int sBh,
    float* __restrict__ outF, bf16* __restrict__ outB,
    int ldc, int sCbm, int sCbn, long long sCb, int sCh,
    const float* __restrict__ bias, float alpha, int gelu,
    const float* __restrict__ res1, const float* __restrict__ res2,
    int M, int N, int K, int H) {
#if HAVE_ASYNC_LDS
  __shared__ __align__(16) bf16 As[2][TM][TK];  // double-buffered
  __shared__ __align__(16) bf16 Bs[2][TN][TK];
#else
  __shared__ __align__(16) bf16 As[1][TM][TK];
  __shared__ __align__(16) bf16 Bs[1][TN][TK];
#endif

  const int tid  = threadIdx.x;
  const int lane = tid & 31;
  const int wave = tid >> 5;
  const int wm = (wave & 3) * 32;   // wave M offset in block tile
  const int wn = (wave >> 2) * 64;  // wave N offset in block tile

  const int z  = blockIdx.z;
  const int bb = z / H;
  const int hh = z % H;
  const long long offA = (long long)bb * sAb + (long long)hh * sAh;
  const long long offB = (long long)bb * sBb + (long long)hh * sBh;
  const long long offC = (long long)bb * sCb + (long long)hh * sCh;

  const int blockM = blockIdx.y * TM;
  const int blockN = blockIdx.x * TN;

  v8f acc[2][4] = {};

  // ISA fragment lane decomposition (CDNA5 ISA 7.12.2)
  const int lm   = lane & 15;         // A row / B col within fragment
  const int lk8  = (lane >> 4) * 8;   // A K base: 0 or 8
  const int lk16 = (lane >> 4) * 16;  // unused; B also uses lk8 as its K base

#if HAVE_ASYNC_LDS
  // -------- async double-buffered pipeline ---------------------------------
  // Each wave issues 8 async b128 instructions per stage (4 for A, 4 for B).
  // OOB rows/cols are address-clamped; their results are never stored.
#define ISSUE_STAGE(p, k0v)                                                  \
  do {                                                                       \
    for (int c = tid; c < (TM * TK) / 8; c += 256) {                         \
      int m = c >> 3, kc = (c & 7) * 8;                                      \
      int gm = blockM + m;                                                   \
      if (gm >= M) gm = M - 1;                                               \
      __builtin_amdgcn_global_load_async_to_lds_b128(                        \
          (as1_v4i*)&A[offA + (long long)gm * sAm + (k0v) + kc],             \
          (as3_v4i*)&As[p][m][kc], 0, 0);                                    \
    }                                                                        \
    for (int c = tid; c < (TN * TK) / 8; c += 256) {                         \
      int n = c >> 3, kc = (c & 7) * 8;                                      \
      int gn = blockN + n;                                                   \
      if (gn >= N) gn = N - 1;                                               \
      __builtin_amdgcn_global_load_async_to_lds_b128(                        \
          (as1_v4i*)&Bt[offB + (long long)gn * sBn + (k0v) + kc],            \
          (as3_v4i*)&Bs[p][n][kc], 0, 0);                                    \
    }                                                                        \
  } while (0)

  const int nst = K / TK;
  ISSUE_STAGE(0, 0);
  for (int i = 0; i < nst; ++i) {
    const int p = i & 1;
    if (i + 1 < nst) {
      ISSUE_STAGE(p ^ 1, (i + 1) * TK);
      WAIT_ASYNC(8);   // stage i complete; stage i+1 still in flight
    } else {
      WAIT_ASYNC(0);
    }
    __syncthreads();   // all waves' loads into buffer p are visible
    COMPUTE_STAGE(As[p], Bs[p]);
    __syncthreads();   // all waves done reading p before it is overwritten
  }
#undef ISSUE_STAGE
#else
  // -------- fallback: b128 load -> ds_store staging ------------------------
  for (int k0 = 0; k0 < K; k0 += TK) {
    for (int c = tid; c < (TM * TK) / 8; c += 256) {
      int m = c >> 3, kc = (c & 7) * 8;
      int gm = blockM + m;
      v8bf v = zero8();
      if (gm < M) v = *(const v8bf*)&A[offA + (long long)gm * sAm + k0 + kc];
      *(v8bf*)&As[0][m][kc] = v;
    }
    for (int c = tid; c < (TN * TK) / 8; c += 256) {
      int n = c >> 3, kc = (c & 7) * 8;
      int gn = blockN + n;
      v8bf v = zero8();
      if (gn < N) v = *(const v8bf*)&Bt[offB + (long long)gn * sBn + k0 + kc];
      *(v8bf*)&Bs[0][n][kc] = v;
    }
    if (k0 + TK < K) {
      int pr = tid & 127;
      int gm = blockM + pr;
      if (gm < M)
        __builtin_prefetch(&A[offA + (long long)gm * sAm + k0 + TK], 0, 1);
      int gn = blockN + pr;
      if (gn < N)
        __builtin_prefetch(&Bt[offB + (long long)gn * sBn + k0 + TK], 0, 1);
    }
    __syncthreads();
    COMPUTE_STAGE(As[0], Bs[0]);
    __syncthreads();
  }
#endif

  // ---- epilogue (C layout: row = (lane>>4)*8 + r, col = lane&15) ---------
  const int cn0 = lane & 15;
  const int cm0 = (lane >> 4) * 8;
  for (int fm = 0; fm < 2; ++fm) {
    for (int fn = 0; fn < 4; ++fn) {
      const int gmb = blockM + wm + fm * 16 + cm0;
      const int gn  = blockN + wn + fn * 16 + cn0;
      if (gn >= N) continue;
      float v[8];
      for (int r = 0; r < 8; ++r) {
        float t = acc[fm][fn][r];
        if (bias) t += bias[gn];
        t *= alpha;
        if (gelu) t = 0.5f * t * (1.0f + erff(t * 0.70710678118654752f));
        v[r] = t;
      }
      if (outF) {
        for (int r = 0; r < 8; ++r) {
          if (gmb + r < M) {
            long long ci = offC + (long long)(gmb + r) * ldc + gn;
            float t = v[r];
            if (res1) t += res1[ci];
            if (res2) t += res2[ci];
            outF[ci] = t;
          }
        }
      }
      if (outB) {
        if (sCbm == 1) {
          // transposed layout: 8 consecutive rows -> one b128 store
          if (gmb + 8 <= M) {
            v8bf pk;
            for (int r = 0; r < 8; ++r) pk[r] = f2bf(v[r]);
            *(v8bf*)&outB[offC + (long long)gn * sCbn + gmb] = pk;
          }
        } else {
          for (int r = 0; r < 8; ++r)
            if (gmb + r < M)
              outB[offC + (long long)(gmb + r) * sCbm +
                   (long long)gn * sCbn] = f2bf(v[r]);
        }
      }
    }
  }
}

extern "C" void kernel_launch(void* const* d_in, const int* in_sizes, int n_in,
                              void* d_out, int out_size, void* d_ws,
                              size_t ws_size, hipStream_t stream) {
  (void)in_sizes; (void)n_in; (void)out_size; (void)ws_size;

  const float* x    = (const float*)d_in[0];
  const float* data = (const float*)d_in[1];
  const float* Wq   = (const float*)d_in[2];
  const float* bq   = (const float*)d_in[3];
  const float* Wk   = (const float*)d_in[4];
  const float* bk   = (const float*)d_in[5];
  const float* Wv   = (const float*)d_in[6];
  const float* bv   = (const float*)d_in[7];
  const float* W1   = (const float*)d_in[8];
  const float* b1   = (const float*)d_in[9];
  const float* W2   = (const float*)d_in[10];
  const float* b2   = (const float*)d_in[11];
  float* out = (float*)d_out;

  const int Bb = 4, S = 2048, C = 1024, Hh = 16, D = 64, MH = 1024;
  const int BS = Bb * S;                       // 8192 rows
  const long long NACT = (long long)BS * C;    // 8,388,608
  const long long NW   = (long long)C * C;     // 1,048,576
  const float SCALE = 0.125f;                  // 64^-0.5

  // --- workspace carve-out (256B aligned), ~95 MB used -------------------
  char* ws = (char*)d_ws;
  size_t off = 0;
  auto alloc = [&](size_t bytes) -> char* {
    char* p = ws + off;
    off += (bytes + 255) & ~(size_t)255;
    return p;
  };
  bf16* xb  = (bf16*)alloc(NACT * 2);          // x, bf16 [BS][C]
  bf16* db  = (bf16*)alloc(NACT * 2);          // data, bf16 [BS][C]
  bf16* wqT = (bf16*)alloc(NW * 2);            // weights, bf16 [N][K]
  bf16* wkT = (bf16*)alloc(NW * 2);
  bf16* wvT = (bf16*)alloc(NW * 2);
  bf16* w1T = (bf16*)alloc(NW * 2);
  bf16* w2T = (bf16*)alloc(NW * 2);
  bf16* qb  = (bf16*)alloc(NACT * 2);          // q, bf16 [BS][C]
  bf16* kbT = (bf16*)alloc(NACT * 2);          // k^T, bf16 [C][BS] (contig w/ vbT)
  bf16* vbT = (bf16*)alloc(NACT * 2);          // v^T, bf16 [C][BS]
  bf16* kvT = (bf16*)alloc((long long)Bb * Hh * D * D * 2);  // kv^T [z][e][d]
  // Aliases over dead buffers (kernels serialize on `stream`):
  float* attnF = (float*)kbT;  // 32 MB f32 over kbT+vbT (dead after kv GEMM)
  bf16*  attnB = db;           // db dead after V projection
  bf16*  hb    = qb;           // qb dead after attn GEMM
  (void)vbT;

  // --- conversions --------------------------------------------------------
  auto cvt = [&](const float* src, bf16* dst, long long n) {
    int blocks = (int)((n + 255) / 256);
    cvt_f32_bf16_kernel<<<blocks, 256, 0, stream>>>(src, dst, (int)n);
  };
  cvt(x, xb, NACT);
  cvt(data, db, NACT);
  dim3 gT(C / 32, C / 32, 1);
  cvtT_f32_bf16_kernel<<<gT, 256, 0, stream>>>(Wq, wqT, C, C);
  cvtT_f32_bf16_kernel<<<gT, 256, 0, stream>>>(Wk, wkT, C, C);
  cvtT_f32_bf16_kernel<<<gT, 256, 0, stream>>>(Wv, wvT, C, C);
  cvtT_f32_bf16_kernel<<<gT, 256, 0, stream>>>(W1, w1T, C, MH);
  cvtT_f32_bf16_kernel<<<gT, 256, 0, stream>>>(W2, w2T, MH, C);

  // --- 1-3) Q/K/V projections: [8192,1024] @ [1024,1024] ------------------
  dim3 gproj(C / TN, BS / TM, 1);
  // q = (x@Wq + bq)*SCALE -> qb [BS][C]
  gemm_bf16_kernel<<<gproj, 256, 0, stream>>>(
      xb, C, 0, 0, wqT, C, 0, 0,
      nullptr, qb, C, C, 1, 0, 0, bq, SCALE, 0, nullptr, nullptr,
      BS, C, C, 1);
  // k = data@Wk + bk -> kbT [C][BS]  (transposed store, vectorized)
  gemm_bf16_kernel<<<gproj, 256, 0, stream>>>(
      db, C, 0, 0, wkT, C, 0, 0,
      nullptr, kbT, C, 1, BS, 0, 0, bk, 1.0f, 0, nullptr, nullptr,
      BS, C, C, 1);
  // v = data@Wv + bv -> vbT [C][BS]
  gemm_bf16_kernel<<<gproj, 256, 0, stream>>>(
      db, C, 0, 0, wvT, C, 0, 0,
      nullptr, vbT, C, 1, BS, 0, 0, bv, 1.0f, 0, nullptr, nullptr,
      BS, C, C, 1);

  // --- 4) kv[b,h] = k^T @ v : 64x64, K=2048, batch=64 ---------------------
  //   A[d,s] = kbT[(h*64+d)*BS + b*S + s]   (K-contiguous)
  //   Bt[e,s] = vbT[(h*64+e)*BS + b*S + s]  (K-contiguous)
  //   out kvT[z][e][d] (transposed store)
  dim3 gkv(1, 1, Bb * Hh);
  gemm_bf16_kernel<<<gkv, 256, 0, stream>>>(
      kbT, BS, S, D * BS, vbT, BS, S, D * BS,
      nullptr, kvT, D, 1, D, (long long)Hh * D * D, D * D,
      nullptr, 1.0f, 0, nullptr, nullptr,
      D, D, S, Hh);

  // --- 5) attn[b,h] = q @ kv : 2048x64, K=64, batch=64 --------------------
  //   A = qb per head (sAm=C, K-contig), Bt = kvT[z] (sBn=D, K-contig)
  dim3 gattn(1, S / TM, Bb * Hh);
  gemm_bf16_kernel<<<gattn, 256, 0, stream>>>(
      qb, C, (long long)S * C, D, kvT, D, (long long)Hh * D * D, D * D,
      attnF, attnB, C, C, 1, (long long)S * C, D,
      nullptr, 1.0f, 0, nullptr, nullptr,
      S, D, D, Hh);

  // --- 6) h = gelu(attn @ W1 + b1) : [8192,1024] --------------------------
  dim3 gmlp(MH / TN, BS / TM, 1);
  gemm_bf16_kernel<<<gmlp, 256, 0, stream>>>(
      attnB, C, 0, 0, w1T, C, 0, 0,
      nullptr, hb, MH, MH, 1, 0, 0, b1, 1.0f, 1, nullptr, nullptr,
      BS, MH, C, 1);

  // --- 7) out = x + attn + (h @ W2 + b2) : [8192,1024] --------------------
  dim3 gout(C / TN, BS / TM, 1);
  gemm_bf16_kernel<<<gout, 256, 0, stream>>>(
      hb, MH, 0, 0, w2T, MH, 0, 0,
      out, nullptr, C, 0, 0, 0, 0, b2, 1.0f, 0, x, attnF,
      BS, C, MH, 1);
}
